// FusionParticleTransformer_4569845203040
// MI455X (gfx1250) — matrix operand order repaired and checked by reading, not compiled, not run on hardware
//
#include <hip/hip_runtime.h>
#include <hip/hip_bf16.h>
#include <stdint.h>

// ---------------------------------------------------------------------------
// Particle Transformer forward for MI455X (gfx1250), wave32 + WMMA f16.
// All GEMMs use v_wmma_f32_16x16x32_f16; Q/K stored with inner stride 32
// (zero-padded) so attention fragment loads are branch-free; V stored
// transposed so a@V B-fragments are contiguous. Attention bias is stored
// transposed (b,h,key,query) WITH the key mask folded in, so the attention
// epilogue is pure vectorized adds. Pair-MLP 64x64 layer is WMMA through
// LDS. Whole working set (~80MB) is L2-resident.
// ---------------------------------------------------------------------------

typedef _Float16 half_t;
typedef __attribute__((ext_vector_type(16))) _Float16 v16h;
typedef __attribute__((ext_vector_type(8)))  _Float16 v8h;
typedef __attribute__((ext_vector_type(8)))  float    v8f;

#define B_   64
#define N_   128
#define CIN_ 17
#define D_   128
#define H_   8
#define HD_  16
#define HDP  32    // padded head-dim stride (upper 16 halves are zero)
#define FF_  512
#define SKV  144   // padded KV sequence stride (>=129, keeps alignment)
#define PI_F 3.14159265358979323846f

union FragU { v16h v; v8h h[2]; };

__device__ inline v8f wmma_f16(v16h a, v16h b, v8f c) {
  // emits v_wmma_f32_16x16x32_f16
  return __builtin_amdgcn_wmma_f32_16x16x32_f16(false, a, false, b, (short)0, c, false, false);
}
__device__ inline v8f zero_v8f(){ v8f z; for (int q=0;q<8;q++) z[q]=0.f; return z; }
__device__ inline float gelu_f(float x){ return 0.5f*x*(1.0f+erff(x*0.70710678118654752f)); }

// ------------------------- zero f16 buffer ---------------------------------
__global__ void k_zero16(half_t* __restrict__ p, int n){
  int i = blockIdx.x*256 + threadIdx.x;
  if (i < n) p[i] = (half_t)0.f;
}

// ------------------------- generic LayerNorm -------------------------------
template<int W, bool SRC16, bool OUTF32>
__global__ void __launch_bounds__(128)
k_ln(const void* __restrict__ src, const float* __restrict__ g, const float* __restrict__ bta,
     half_t* __restrict__ dst16, float* __restrict__ dst32)
{
  int row = blockIdx.x, tid = threadIdx.x;
  constexpr int NV = W / 128;
  __shared__ float s1[128], s2[128];
  float v[NV]; float ps = 0.f, ps2 = 0.f;
  for (int i = 0; i < NV; i++) {
    int f = i*128 + tid;
    float x = SRC16 ? (float)((const half_t*)src)[(size_t)row*W + f]
                    : ((const float*)src)[(size_t)row*W + f];
    v[i] = x; ps += x; ps2 += x*x;
  }
  s1[tid] = ps; s2[tid] = ps2; __syncthreads();
  for (int off = 64; off > 0; off >>= 1) {
    if (tid < off) { s1[tid] += s1[tid+off]; s2[tid] += s2[tid+off]; }
    __syncthreads();
  }
  float mu = s1[0] / (float)W;
  float var = s2[0] / (float)W - mu*mu;
  float rstd = rsqrtf(var + 1e-5f);
  for (int i = 0; i < NV; i++) {
    int f = i*128 + tid;
    float y = (v[i]-mu)*rstd*g[f] + bta[f];
    dst16[(size_t)row*W + f] = (half_t)y;
    if constexpr (OUTF32) dst32[(size_t)row*W + f] = y;
  }
}

// ------------------------- f32 -> f16 weight convert -----------------------
__global__ void k_w2h(const float* __restrict__ src, half_t* __restrict__ dst, int n){
  int i = blockIdx.x*256 + threadIdx.x;
  if (i < n) dst[i] = (half_t)src[i];
}

// ------------------------- WMMA GEMM: D = A * W^T + bias -------------------
// A: MxK f16 row-major (lda=K). W: NxK f16 row-major (so B[k][n]=W[n][k]).
// MODE 0: f32 out.  MODE 1: f16 gelu out.  MODE 2: f32 gelu out.
// MODE 3: QKV scatter (q scaled by HD^-0.5; Q/K padded layout, V transposed).
template<int MODE>
__global__ void __launch_bounds__(256)
k_gemm(const half_t* __restrict__ A, int lda,
       const half_t* __restrict__ W, int K, int N, int M,
       const float* __restrict__ bias,
       float* __restrict__ outF32, half_t* __restrict__ outF16, int ldo,
       int nBase, half_t* __restrict__ Qb, half_t* __restrict__ Kb, half_t* __restrict__ Vt)
{
  int wv = threadIdx.x >> 5, lane = threadIdx.x & 31;
  int lh = lane & 15; bool hiL = lane >= 16;
  int rbase = blockIdx.x*128 + wv*16;
  if (rbase >= M) return;
  int cbase = blockIdx.y*64;
  const half_t* arow = A + (size_t)(rbase + lh)*lda;
  v8f acc[4];
  #pragma unroll
  for (int j = 0; j < 4; j++) acc[j] = zero_v8f();
  for (int k0 = 0; k0 < K; k0 += 32) {
    __builtin_prefetch(arow + k0 + 64, 0, 1);   // global_prefetch_b8
    FragU af;
    int c0 = k0 + (hiL ? 16 : 0);
    af.h[0] = *(const v8h*)(arow + c0);
    af.h[1] = *(const v8h*)(arow + c0 + 8);
    int bk = k0 + (hiL ? 16 : 0);
    #pragma unroll
    for (int j = 0; j < 4; j++) {
      // 32B-aligned: K multiple of 32, bk multiple of 16
      v16h bf = *(const v16h*)(W + (size_t)(cbase + j*16 + lh)*K + bk);
      acc[j] = wmma_f16(af.v, bf, acc[j]);
    }
  }
  int rowoff = hiL ? 8 : 0;
  #pragma unroll
  for (int j = 0; j < 4; j++) {
    int col = cbase + j*16 + lh;
    float bv = bias ? bias[col] : 0.f;
    #pragma unroll
    for (int r = 0; r < 8; r++) {
      int row = rbase + r + rowoff;
      float val = acc[j][r] + bv;
      if (MODE == 0) {
        outF32[(size_t)row*ldo + col] = val;
      } else if (MODE == 1) {
        outF16[(size_t)row*ldo + col] = (half_t)gelu_f(val);
      } else if (MODE == 2) {
        outF32[(size_t)row*ldo + col] = gelu_f(val);
      } else {
        int n2 = col + nBase;
        int t = row / B_, b = row - t*B_;
        int h = (n2 >> 4) & 7, d = n2 & 15, which = n2 >> 7;
        if (which == 0)      Qb[(((size_t)b*H_ + h)*N_  + t)*HDP + d] = (half_t)(val * 0.25f);
        else if (which == 1) Kb[(((size_t)b*H_ + h)*SKV + t)*HDP + d] = (half_t)val;
        else                 Vt[(((size_t)b*H_ + h)*HD_ + d)*SKV + t] = (half_t)val;
      }
    }
  }
}

// ------------------------- fused attention per (b,h) -----------------------
// biasT[b,h,key,query] already contains the key mask (-1e9 on padded keys).
__global__ void __launch_bounds__(256)
k_attn(const half_t* __restrict__ Q, const half_t* __restrict__ Kb,
       const half_t* __restrict__ Vt, const float* __restrict__ biasT,
       half_t* __restrict__ O16)
{
  __shared__ half_t aLDS[8][16][136];
  int bh = blockIdx.x, b = bh >> 3, h = bh & 7;
  int wv = threadIdx.x >> 5, lane = threadIdx.x & 31;
  int lh = lane & 15; bool hiL = lane >= 16;
  int rbase = wv*16, rowoff = hiL ? 8 : 0;
  const half_t* Qp = Q  + ((size_t)bh*N_ )*HDP;
  const half_t* Kp = Kb + ((size_t)bh*SKV)*HDP;
  const half_t* Vp = Vt + ((size_t)bh*HD_)*SKV;

  // S = q*k^T.  HD=16 < K=32: upper K-halves read the zero pad -> every lane
  // issues an unconditional aligned load (no EXEC juggling).
  FragU qf;
  {
    const half_t* qr = Qp + (size_t)(rbase + lh)*HDP + (hiL ? 16 : 0);
    qf.h[0] = *(const v8h*)(qr);
    qf.h[1] = *(const v8h*)(qr + 8);    // hi lanes: pad -> zeros
  }
  v8f sacc[8];
  #pragma unroll
  for (int ct = 0; ct < 8; ct++) {
    v16h kf = *(const v16h*)(Kp + (size_t)(ct*16 + lh)*HDP + (hiL ? 16 : 0));
    sacc[ct] = wmma_f16(qf.v, kf, zero_v8f());
  }
  // masked bias add: transposed layout -> 8 contiguous floats per column tile
  const float* bp = biasT + ((size_t)bh*N_)*N_;
  #pragma unroll
  for (int ct = 0; ct < 8; ct++) {
    const float* bb = bp + (size_t)(ct*16 + lh)*N_ + rbase + rowoff;  // 32B aligned
    float4 b0 = *(const float4*)(bb);
    float4 b1 = *(const float4*)(bb + 4);
    sacc[ct][0] += b0.x; sacc[ct][1] += b0.y; sacc[ct][2] += b0.z; sacc[ct][3] += b0.w;
    sacc[ct][4] += b1.x; sacc[ct][5] += b1.y; sacc[ct][6] += b1.z; sacc[ct][7] += b1.w;
  }
  // softmax per row (16-lane-half butterfly: masks 1,2,4,8 stay inside half)
  #pragma unroll
  for (int r = 0; r < 8; r++) {
    float mx = -3.4e38f;
    #pragma unroll
    for (int ct = 0; ct < 8; ct++) mx = fmaxf(mx, sacc[ct][r]);
    for (int m = 1; m < 16; m <<= 1) mx = fmaxf(mx, __shfl_xor(mx, m, 32));
    float sum = 0.f;
    #pragma unroll
    for (int ct = 0; ct < 8; ct++) { float e = __expf(sacc[ct][r]-mx); sacc[ct][r]=e; sum+=e; }
    for (int m = 1; m < 16; m <<= 1) sum += __shfl_xor(sum, m, 32);
    float inv = 1.0f / sum;
    #pragma unroll
    for (int ct = 0; ct < 8; ct++) sacc[ct][r] *= inv;
  }
  // C-layout -> A-layout via LDS (wave-private rows)
  #pragma unroll
  for (int ct = 0; ct < 8; ct++)
    #pragma unroll
    for (int r = 0; r < 8; r++)
      aLDS[wv][r + rowoff][ct*16 + lh] = (half_t)sacc[ct][r];
  __syncthreads();
  // O = a * V  (K = 128, 4 k-steps; B-frags contiguous thanks to Vt layout)
  v8f oacc = zero_v8f();
  #pragma unroll
  for (int ks = 0; ks < 4; ks++) {
    FragU af;
    int c0 = ks*32 + (hiL ? 16 : 0);
    af.h[0] = *(const v8h*)&aLDS[wv][lh][c0];
    af.h[1] = *(const v8h*)&aLDS[wv][lh][c0 + 8];
    FragU vf;
    int k0 = ks*32 + (hiL ? 16 : 0);
    const half_t* vr = Vp + (size_t)lh*SKV + k0;
    vf.h[0] = *(const v8h*)vr; vf.h[1] = *(const v8h*)(vr + 8);
    oacc = wmma_f16(af.v, vf.v, oacc);
  }
  #pragma unroll
  for (int r = 0; r < 8; r++) {
    int t = rbase + r + rowoff;
    O16[((size_t)t*B_ + b)*D_ + h*HD_ + lh] = (half_t)oacc[r];
  }
}

// -------------- head-scale permute (tbhd,h->tbdh) + LN2 + residual ---------
__global__ void __launch_bounds__(128)
k_caln(const float* __restrict__ T0, const float* __restrict__ res,
       const float* __restrict__ ca, const float* __restrict__ g,
       const float* __restrict__ bta, float* __restrict__ out)
{
  int row = blockIdx.x, tid = threadIdx.x;
  __shared__ float s1[128], s2[128];
  int d = tid >> 3, h = tid & 7;              // out feature f = d*H + h
  float x = T0[(size_t)row*D_ + h*HD_ + d] * ca[h];
  s1[tid] = x; s2[tid] = x*x; __syncthreads();
  for (int off = 64; off > 0; off >>= 1) {
    if (tid < off) { s1[tid]+=s1[tid+off]; s2[tid]+=s2[tid+off]; }
    __syncthreads();
  }
  float mu = s1[0]/128.f, var = s2[0]/128.f - mu*mu, rstd = rsqrtf(var + 1e-5f);
  out[(size_t)row*D_ + tid] = (x-mu)*rstd*g[tid] + bta[tid] + res[(size_t)row*D_ + tid];
}

// ------------------------- h = T0 + wr * h ---------------------------------
__global__ void k_addwr(const float* __restrict__ T0, const float* __restrict__ wr,
                        float* __restrict__ hio, int total){
  int i = blockIdx.x*256 + threadIdx.x;
  if (i < total) { int f = i & 127; hio[i] = T0[i] + wr[f]*hio[i]; }
}

// -------------- embed layer 0: batchnorm + LN(17) + 17->128 + gelu ---------
__global__ void __launch_bounds__(128)
k_embed0(const float* __restrict__ x,
         const float* bng, const float* bnb, const float* bnm, const float* bnv,
         const float* lg, const float* lb, const float* w0, const float* b0,
         float* __restrict__ hout)
{
  int m = blockIdx.x, t = m / B_, b = m - t*B_, tid = threadIdx.x;
  __shared__ float s17[32]; __shared__ float stats[2];
  if (tid < CIN_) {
    float xv = x[((size_t)b*CIN_ + tid)*N_ + t];
    s17[tid] = (xv - bnm[tid]) * rsqrtf(bnv[tid] + 1e-5f) * bng[tid] + bnb[tid];
  }
  __syncthreads();
  if (tid == 0) {
    float s = 0.f, s2 = 0.f;
    for (int c = 0; c < CIN_; c++) { s += s17[c]; s2 += s17[c]*s17[c]; }
    float mu = s/(float)CIN_;
    stats[0] = mu; stats[1] = rsqrtf(s2/(float)CIN_ - mu*mu + 1e-5f);
  }
  __syncthreads();
  float mu = stats[0], rstd = stats[1];
  float acc = b0[tid];
  for (int c = 0; c < CIN_; c++)
    acc += ((s17[c]-mu)*rstd*lg[c] + lb[c]) * w0[tid*CIN_ + c];
  hout[(size_t)m*D_ + tid] = gelu_f(acc);
}

// ------------------- per-particle kinematics precompute --------------------
__global__ void k_prep(const float* __restrict__ v, float* pt, float* rap, float* phi){
  int i = blockIdx.x*256 + threadIdx.x;
  if (i >= B_*N_) return;
  int b = i / N_, t = i - b*N_;
  float px = v[((size_t)b*4+0)*N_+t], py = v[((size_t)b*4+1)*N_+t];
  float pz = v[((size_t)b*4+2)*N_+t], E  = v[((size_t)b*4+3)*N_+t];
  pt[i]  = sqrtf(px*px + py*py);
  rap[i] = 0.5f*logf(1.0f + 2.0f*pz/fmaxf(E - pz, 1e-20f));
  phi[i] = atan2f(py, px);
}

// ------------- fused pair features + MLP (64x64 layer via WMMA) ------------
// Thread decode: query index i is FASTEST so the transposed masked-bias
// write biasT[b,h,key=j,query=i] stays coalesced.
__global__ void __launch_bounds__(256)
k_pairbias(const float* __restrict__ v, const float* __restrict__ pt,
           const float* __restrict__ rap, const float* __restrict__ phi,
           const float* pw0, const float* pb0, const float* pw1, const float* pb1,
           const float* pw2, const float* pb2, const uint8_t* __restrict__ mask,
           float* __restrict__ biasT)
{
  __shared__ half_t u1[256][72];     // activations, reused for layer-2 out
  __shared__ half_t w1s[64][72];     // 64x64 weight, f16
  __shared__ float  w0s[64*4 + 64];
  __shared__ float  w2s[8*64 + 8];
  __shared__ float  pb1s[64];
  int tid = threadIdx.x;
  for (int w = tid; w < 64*4;  w += 256) w0s[w] = pw0[w];
  for (int w = tid; w < 64;    w += 256) w0s[256+w] = pb0[w];
  for (int w = tid; w < 64*64; w += 256) w1s[w>>6][w&63] = (half_t)pw1[w];
  for (int w = tid; w < 64;    w += 256) pb1s[w] = pb1[w];
  for (int w = tid; w < 8*64;  w += 256) w2s[w] = pw2[w];
  for (int w = tid; w < 8;     w += 256) w2s[512+w] = pb2[w];

  int p = blockIdx.x*256 + tid;
  int b = p >> 14, rr = p & 16383;
  int j = rr >> 7, i = rr & 127;      // j = key (slow), i = query (fast)
  const float* vb = v + (size_t)b*4*N_;
  float pxi=vb[i], pyi=vb[N_+i], pzi=vb[2*N_+i], Ei=vb[3*N_+i];
  float pxj=vb[j], pyj=vb[N_+j], pzj=vb[2*N_+j], Ej=vb[3*N_+j];
  int ib = b*N_+i, jb = b*N_+j;
  float pti = pt[ib], ptj = pt[jb];
  float drap = rap[ib]-rap[jb];
  float dphi = fmodf(phi[ib]-phi[jb] + PI_F, 2.0f*PI_F);
  if (dphi < 0.f) dphi += 2.0f*PI_F;
  dphi -= PI_F;
  float delta = sqrtf(drap*drap + dphi*dphi);
  float f2 = logf(fmaxf(delta, 1e-8f));
  float ptmin = fminf(pti, ptj);
  float f0 = logf(fmaxf(ptmin*delta, 1e-8f));
  float f1 = logf(fmaxf(ptmin/fmaxf(pti+ptj, 1e-8f), 1e-8f));
  float pxs=pxi+pxj, pys=pyi+pyj, pzs=pzi+pzj, Es=Ei+Ej;
  float f3 = logf(fmaxf(Es*Es - (pxs*pxs + pys*pys + pzs*pzs), 1e-8f));
  bool padk = (mask[b*N_ + j] == 0);
  __syncthreads();

  // layer 1: 4 -> 64 (VALU), gelu, f16 into LDS
  for (int k = 0; k < 64; k++) {
    float a = w0s[256+k] + f0*w0s[k*4] + f1*w0s[k*4+1] + f2*w0s[k*4+2] + f3*w0s[k*4+3];
    u1[tid][k] = (half_t)gelu_f(a);
  }
  __syncthreads();

  // layer 2: 64x64 via WMMA; each wave owns its 32 LDS rows
  int wv = tid >> 5, lane = tid & 31, lh = lane & 15; bool hiL = lane >= 16;
  int rowoff = hiL ? 8 : 0;
  v8f acc[2][4];
  #pragma unroll
  for (int rt = 0; rt < 2; rt++)
    #pragma unroll
    for (int jj = 0; jj < 4; jj++) acc[rt][jj] = zero_v8f();
  #pragma unroll
  for (int ks = 0; ks < 2; ks++) {
    int c0 = ks*32 + (hiL ? 16 : 0);
    int bk = ks*32 + (hiL ? 16 : 0);
    FragU bf[4];
    #pragma unroll
    for (int jj = 0; jj < 4; jj++) {
      bf[jj].h[0] = *(const v8h*)&w1s[jj*16 + lh][bk];
      bf[jj].h[1] = *(const v8h*)&w1s[jj*16 + lh][bk + 8];
    }
    #pragma unroll
    for (int rt = 0; rt < 2; rt++) {
      int row = wv*32 + rt*16 + lh;
      FragU af;
      af.h[0] = *(const v8h*)&u1[row][c0];
      af.h[1] = *(const v8h*)&u1[row][c0 + 8];
      #pragma unroll
      for (int jj = 0; jj < 4; jj++) acc[rt][jj] = wmma_f16(af.v, bf[jj].v, acc[rt][jj]);
    }
  }
  #pragma unroll
  for (int rt = 0; rt < 2; rt++)
    #pragma unroll
    for (int jj = 0; jj < 4; jj++) {
      int col = jj*16 + lh;
      float bv = pb1s[col];
      #pragma unroll
      for (int r = 0; r < 8; r++)
        u1[wv*32 + rt*16 + r + rowoff][col] = (half_t)gelu_f(acc[rt][jj][r] + bv);
    }
  __syncthreads();

  // layer 3: 64 -> 8 (VALU); fold key mask; scatter transposed (coalesced)
  float o[8];
  for (int hh = 0; hh < 8; hh++) o[hh] = w2s[512 + hh];
  for (int k = 0; k < 64; k++) {
    float u = (float)u1[tid][k];
    for (int hh = 0; hh < 8; hh++) o[hh] += u * w2s[hh*64 + k];
  }
  for (int hh = 0; hh < 8; hh++)
    biasT[(((size_t)b*H_ + hh)*N_ + j)*N_ + i] = padk ? -1e9f : o[hh];
}

// ----------------------- class-attention helpers ---------------------------
__global__ void k_bcast_cls(const float* __restrict__ cls_token, float* __restrict__ xcls){
  int i = blockIdx.x*256 + threadIdx.x;
  if (i < B_*D_) xcls[i] = cls_token[i & 127];
}

__global__ void k_cat_cls(const float* __restrict__ xcls, const float* __restrict__ h,
                          float* __restrict__ U){
  int idx = blockIdx.x*256 + threadIdx.x;
  if (idx >= (N_+1)*B_*D_) return;
  int f = idx & 127, m = idx >> 7;
  int s = m / B_, b = m - s*B_;
  U[idx] = (s == 0) ? xcls[b*D_ + f] : h[((size_t)(s-1)*B_ + b)*D_ + f];
}

__global__ void __launch_bounds__(128)
k_qproj_cls(const float* __restrict__ xcls, const float* __restrict__ qkvw,
            const float* __restrict__ qkvb, half_t* __restrict__ Qc){
  int b = blockIdx.x, n = threadIdx.x;
  const float* wr = qkvw + (size_t)n*D_;
  const float* xr = xcls + (size_t)b*D_;
  float acc = qkvb[n];
  for (int c = 0; c < D_; c++) acc += xr[c]*wr[c];
  Qc[(size_t)b*D_ + n] = (half_t)(acc * 0.25f);
}

__global__ void __launch_bounds__(256)
k_attn_cls(const half_t* __restrict__ Qc, const half_t* __restrict__ Kb,
           const half_t* __restrict__ Vt, const uint8_t* __restrict__ mask,
           half_t* __restrict__ O16)
{
  __shared__ float sa[132]; __shared__ float red[2];
  int bh = blockIdx.x, b = bh >> 3, h = bh & 7, tid = threadIdx.x;
  const half_t* Kp = Kb + (size_t)bh*SKV*HDP;
  const half_t* Vp = Vt + (size_t)bh*HD_*SKV;
  if (tid < 129) {
    bool pad = (tid > 0) && (mask[b*N_ + tid-1] == 0);
    float s = 0.f;
    for (int d = 0; d < HD_; d++)
      s += (float)Qc[b*D_ + h*HD_ + d] * (float)Kp[(size_t)tid*HDP + d];
    sa[tid] = pad ? -1e9f : s;
  }
  __syncthreads();
  if (tid == 0) { float mx=-3.4e38f; for (int j=0;j<129;j++) mx=fmaxf(mx,sa[j]); red[0]=mx; }
  __syncthreads();
  if (tid < 129) sa[tid] = __expf(sa[tid] - red[0]);
  __syncthreads();
  if (tid == 0) { float s=0.f; for (int j=0;j<129;j++) s+=sa[j]; red[1]=s; }
  __syncthreads();
  if (tid < HD_) {
    float inv = 1.0f/red[1], o = 0.f;
    for (int j = 0; j < 129; j++) o += sa[j] * (float)Vp[(size_t)tid*SKV + j];
    O16[(size_t)b*D_ + h*HD_ + tid] = (half_t)(o * inv);
  }
}

// ----------------------------- final head ----------------------------------
__global__ void __launch_bounds__(128)
k_head(const float* __restrict__ xcls, const float* __restrict__ g, const float* __restrict__ bta,
       const float* __restrict__ fcw, const float* __restrict__ fcb, float* __restrict__ out){
  __shared__ float s1[128], s2[128], nrm[128];
  int b = blockIdx.x, tid = threadIdx.x;
  float x = xcls[(size_t)b*D_ + tid];
  s1[tid] = x; s2[tid] = x*x; __syncthreads();
  for (int off = 64; off > 0; off >>= 1) {
    if (tid < off) { s1[tid]+=s1[tid+off]; s2[tid]+=s2[tid+off]; }
    __syncthreads();
  }
  float mu = s1[0]/128.f, var = s2[0]/128.f - mu*mu, rstd = rsqrtf(var + 1e-5f);
  nrm[tid] = (x-mu)*rstd*g[tid] + bta[tid];
  __syncthreads();
  if (tid < 10) {
    float acc = fcb[tid];
    for (int c = 0; c < 128; c++) acc += nrm[c]*fcw[tid*128 + c];
    out[b*10 + tid] = acc;
  }
}

// ===========================================================================
extern "C" void kernel_launch(void* const* d_in, const int* in_sizes, int n_in,
                              void* d_out, int out_size, void* d_ws, size_t ws_size,
                              hipStream_t stream) {
  (void)in_sizes; (void)n_in; (void)out_size; (void)ws_size;
  const float* x       = (const float*)d_in[0];
  const float* vv      = (const float*)d_in[1];
  const uint8_t* mask  = (const uint8_t*)d_in[2];
  const float* bn_g=(const float*)d_in[3],  *bn_b=(const float*)d_in[4];
  const float* bn_m=(const float*)d_in[5],  *bn_v=(const float*)d_in[6];
  const float* e_ln_g0=(const float*)d_in[7],  *e_ln_b0=(const float*)d_in[8];
  const float* e_w0=(const float*)d_in[9],     *e_b0=(const float*)d_in[10];
  const float* e_ln_g1=(const float*)d_in[11], *e_ln_b1=(const float*)d_in[12];
  const float* e_w1=(const float*)d_in[13],    *e_b1=(const float*)d_in[14];
  const float* e_ln_g2=(const float*)d_in[15], *e_ln_b2=(const float*)d_in[16];
  const float* e_w2=(const float*)d_in[17],    *e_b2=(const float*)d_in[18];
  const float* p_w0=(const float*)d_in[19], *p_b0=(const float*)d_in[20];
  const float* p_w1=(const float*)d_in[21], *p_b1=(const float*)d_in[22];
  const float* p_w2=(const float*)d_in[23], *p_b2=(const float*)d_in[24];
  const float* b_ln1g=(const float*)d_in[25], *b_ln1b=(const float*)d_in[26];
  const float* b_qkvw=(const float*)d_in[27], *b_qkvb=(const float*)d_in[28];
  const float* b_outw=(const float*)d_in[29], *b_outb=(const float*)d_in[30];
  const float* b_ln2g=(const float*)d_in[31], *b_ln2b=(const float*)d_in[32];
  const float* b_ln3g=(const float*)d_in[33], *b_ln3b=(const float*)d_in[34];
  const float* b_fc1w=(const float*)d_in[35], *b_fc1b=(const float*)d_in[36];
  const float* b_ln4g=(const float*)d_in[37], *b_ln4b=(const float*)d_in[38];
  const float* b_fc2w=(const float*)d_in[39], *b_fc2b=(const float*)d_in[40];
  const float* b_ca=(const float*)d_in[41],   *b_wr=(const float*)d_in[42];
  const float* c_ln1g=(const float*)d_in[43], *c_ln1b=(const float*)d_in[44];
  const float* c_qkvw=(const float*)d_in[45], *c_qkvb=(const float*)d_in[46];
  const float* c_outw=(const float*)d_in[47], *c_outb=(const float*)d_in[48];
  const float* c_ln2g=(const float*)d_in[49], *c_ln2b=(const float*)d_in[50];
  const float* c_ln3g=(const float*)d_in[51], *c_ln3b=(const float*)d_in[52];
  const float* c_fc1w=(const float*)d_in[53], *c_fc1b=(const float*)d_in[54];
  const float* c_ln4g=(const float*)d_in[55], *c_ln4b=(const float*)d_in[56];
  const float* c_fc2w=(const float*)d_in[57], *c_fc2b=(const float*)d_in[58];
  const float* c_ca=(const float*)d_in[59],   *c_wr=(const float*)d_in[60];
  const float* cls_token=(const float*)d_in[61];
  const float* f_ln_g=(const float*)d_in[62], *f_ln_b=(const float*)d_in[63];
  const float* fc_w=(const float*)d_in[64],   *fc_b=(const float*)d_in[65];

  // ---- workspace carving (~80 MB total, whole set L2-resident) ----
  size_t off = 0; char* base = (char*)d_ws;
  auto carve = [&](size_t bytes)->void* {
    void* q = base + off; off += (bytes + 255) & ~(size_t)255; return q;
  };
  float*  h     = (float*) carve((size_t)8192*128*4);
  float*  U     = (float*) carve((size_t)8256*128*4);
  float*  Xf32  = (float*) carve((size_t)8192*128*4);
  half_t* X16   = (half_t*)carve((size_t)8256*512*2);
  half_t* A2    = (half_t*)carve((size_t)8192*512*2);
  float*  T0    = (float*) carve((size_t)8192*128*4);
  half_t* Qb    = (half_t*)carve((size_t)B_*H_*N_*HDP*2);
  half_t* Kb    = (half_t*)carve((size_t)B_*H_*SKV*HDP*2);
  half_t* Vt    = (half_t*)carve((size_t)B_*H_*HD_*SKV*2);
  half_t* O16   = (half_t*)carve((size_t)8192*128*2);
  float*  biasT = (float*) carve((size_t)B_*H_*N_*N_*4);
  half_t* w16   = (half_t*)carve((size_t)65536*2);
  half_t* Qc    = (half_t*)carve((size_t)B_*D_*2);
  float*  xcls  = (float*) carve((size_t)B_*D_*4);
  float*  ptb   = (float*) carve((size_t)B_*N_*4);
  float*  rapb  = (float*) carve((size_t)B_*N_*4);
  float*  phib  = (float*) carve((size_t)B_*N_*4);

  // zero Q/K pad lanes once per launch (scatter only writes d<16, pads stay 0)
  {
    int nQ = B_*H_*N_*HDP, nK = B_*H_*SKV*HDP;
    k_zero16<<<(nQ+255)/256, 256, 0, stream>>>(Qb, nQ);
    k_zero16<<<(nK+255)/256, 256, 0, stream>>>(Kb, nK);
  }

  // ---- pair-feature attention bias (mask folded in, transposed) ----
  k_prep<<<32, 256, 0, stream>>>(vv, ptb, rapb, phib);
  k_pairbias<<<4096, 256, 0, stream>>>(vv, ptb, rapb, phib,
                                       p_w0, p_b0, p_w1, p_b1, p_w2, p_b2,
                                       mask, biasT);

  // ---- embedding MLP ----
  k_embed0<<<8192, 128, 0, stream>>>(x, bn_g, bn_b, bn_m, bn_v,
                                     e_ln_g0, e_ln_b0, e_w0, e_b0, h);
  k_ln<128,false,false><<<8192, 128, 0, stream>>>(h, e_ln_g1, e_ln_b1, X16, nullptr);
  k_w2h<<<256, 256, 0, stream>>>(e_w1, w16, 65536);
  k_gemm<1><<<dim3(64,8), 256, 0, stream>>>(X16, 128, w16, 128, 512, 8192,
      e_b1, nullptr, A2, 512, 0, nullptr, nullptr, nullptr);
  k_ln<512,true,false><<<8192, 128, 0, stream>>>(A2, e_ln_g2, e_ln_b2, X16, nullptr);
  k_w2h<<<256, 256, 0, stream>>>(e_w2, w16, 65536);
  k_gemm<2><<<dim3(64,2), 256, 0, stream>>>(X16, 512, w16, 512, 128, 8192,
      e_b2, h, nullptr, 128, 0, nullptr, nullptr, nullptr);

  // ---- 8 self-attention blocks ----
  for (int l = 0; l < 8; l++) {
    k_ln<128,false,true><<<8192, 128, 0, stream>>>(h, b_ln1g+l*128, b_ln1b+l*128, X16, Xf32);
    k_w2h<<<192, 256, 0, stream>>>(b_qkvw + (size_t)l*384*128, w16, 49152);
    k_gemm<3><<<dim3(64,6), 256, 0, stream>>>(X16, 128, w16, 128, 384, 8192,
        b_qkvb+l*384, nullptr, nullptr, 0, 0, Qb, Kb, Vt);
    k_attn<<<512, 256, 0, stream>>>(Qb, Kb, Vt, biasT, O16);
    k_w2h<<<64, 256, 0, stream>>>(b_outw + (size_t)l*128*128, w16, 16384);
    k_gemm<0><<<dim3(64,2), 256, 0, stream>>>(O16, 128, w16, 128, 128, 8192,
        b_outb+l*128, T0, nullptr, 128, 0, nullptr, nullptr, nullptr);
    k_caln<<<8192, 128, 0, stream>>>(T0, Xf32, b_ca+l*8, b_ln2g+l*128, b_ln2b+l*128, h);
    k_ln<128,false,false><<<8192, 128, 0, stream>>>(h, b_ln3g+l*128, b_ln3b+l*128, X16, nullptr);
    k_w2h<<<256, 256, 0, stream>>>(b_fc1w + (size_t)l*512*128, w16, 65536);
    k_gemm<1><<<dim3(64,8), 256, 0, stream>>>(X16, 128, w16, 128, 512, 8192,
        b_fc1b+l*512, nullptr, A2, 512, 0, nullptr, nullptr, nullptr);
    k_ln<512,true,false><<<8192, 128, 0, stream>>>(A2, b_ln4g+l*512, b_ln4b+l*512, X16, nullptr);
    k_w2h<<<256, 256, 0, stream>>>(b_fc2w + (size_t)l*128*512, w16, 65536);
    k_gemm<0><<<dim3(64,2), 256, 0, stream>>>(X16, 512, w16, 512, 128, 8192,
        b_fc2b+l*128, T0, nullptr, 128, 0, nullptr, nullptr, nullptr);
    k_addwr<<<4096, 256, 0, stream>>>(T0, b_wr+l*128, h, 8192*128);
  }

  // ---- 2 class-attention blocks ----
  k_bcast_cls<<<32, 256, 0, stream>>>(cls_token, xcls);
  for (int l = 0; l < 2; l++) {
    k_cat_cls<<<4128, 256, 0, stream>>>(xcls, h, U);
    k_ln<128,false,false><<<8256, 128, 0, stream>>>(U, c_ln1g+l*128, c_ln1b+l*128, X16, nullptr);
    k_qproj_cls<<<64, 128, 0, stream>>>(xcls, c_qkvw + (size_t)l*384*128, c_qkvb+l*384, Qc);
    k_w2h<<<128, 256, 0, stream>>>(c_qkvw + (size_t)l*384*128 + 128*128, w16, 32768);
    k_gemm<3><<<dim3(65,4), 256, 0, stream>>>(X16, 128, w16, 128, 256, 8256,
        c_qkvb+l*384+128, nullptr, nullptr, 0, 128, Qb, Kb, Vt);
    k_attn_cls<<<512, 256, 0, stream>>>(Qc, Kb, Vt, mask, O16);
    k_w2h<<<64, 256, 0, stream>>>(c_outw + (size_t)l*128*128, w16, 16384);
    k_gemm<0><<<dim3(1,2), 256, 0, stream>>>(O16, 128, w16, 128, 128, 64,
        c_outb+l*128, T0, nullptr, 128, 0, nullptr, nullptr, nullptr);
    k_caln<<<64, 128, 0, stream>>>(T0, xcls, c_ca+l*8, c_ln2g+l*128, c_ln2b+l*128, xcls);
    k_ln<128,false,false><<<64, 128, 0, stream>>>(xcls, c_ln3g+l*128, c_ln3b+l*128, X16, nullptr);
    k_w2h<<<256, 256, 0, stream>>>(c_fc1w + (size_t)l*512*128, w16, 65536);
    k_gemm<1><<<dim3(1,8), 256, 0, stream>>>(X16, 128, w16, 128, 512, 64,
        c_fc1b+l*512, nullptr, A2, 512, 0, nullptr, nullptr, nullptr);
    k_ln<512,true,false><<<64, 128, 0, stream>>>(A2, c_ln4g+l*512, c_ln4b+l*512, X16, nullptr);
    k_w2h<<<256, 256, 0, stream>>>(c_fc2w + (size_t)l*128*512, w16, 65536);
    k_gemm<0><<<dim3(1,2), 256, 0, stream>>>(X16, 512, w16, 512, 128, 64,
        c_fc2b+l*128, T0, nullptr, 128, 0, nullptr, nullptr, nullptr);
    k_addwr<<<32, 256, 0, stream>>>(T0, c_wr+l*128, xcls, 64*128);
  }

  // ---- head ----
  k_head<<<64, 128, 0, stream>>>(xcls, f_ln_g, f_ln_b, fc_w, fc_b, (float*)d_out);
}
